// GeneResModel_two_layer_61847529062922
// MI455X (gfx1250) — matrix operand order repaired and checked by reading, compiled, not run on hardware
//
#include <hip/hip_runtime.h>
#include <hip/hip_bf16.h>
#include <stdint.h>

// ---------------------------------------------------------------------------
// GeneResModel on MI455X (gfx1250, wave32, WMMA)
//   g  = relu(blockdiag(x) . gene_w + gene_b)           [1024, 2000]  (HBM-bound, 200MB x)
//   h1 = relu(BN(g @ W1))     -- b1 cancels inside BN   [1024, 1024]  (bf16 WMMA)
//   h2 = relu(BN(h1 @ W2))    -- b2 cancels inside BN   [1024, 512]   (bf16 WMMA)
//   out= softmax(h2@Wout + bout + g@Wres + bres)        [1024, 10]
// ---------------------------------------------------------------------------

namespace {
constexpr int kB   = 1024;
constexpr int kG   = 2000;
constexpr int kGP  = 2048;   // G padded to multiple of 32 (WMMA K granularity)
constexpr int kGS  = 25;
constexpr int kH1  = 1024;
constexpr int kH2  = 512;
constexpr int kC   = 10;
constexpr float kEps = 1e-5f;
}

typedef __attribute__((ext_vector_type(16))) __bf16 bf16x16;
typedef __attribute__((ext_vector_type(8)))  float  v8f;
typedef __attribute__((ext_vector_type(4)))  float  f32x4;

union FragAB { bf16x16 v; f32x4 f[2]; };

static __device__ __forceinline__ uint16_t f2bf(float x) {
  // round-to-nearest-even fp32 -> bf16
  uint32_t u = __builtin_bit_cast(uint32_t, x);
  u += 0x7FFFu + ((u >> 16) & 1u);
  return (uint16_t)(u >> 16);
}
static __device__ __forceinline__ float bf2f(uint16_t h) {
  uint32_t u = ((uint32_t)h) << 16;
  return __builtin_bit_cast(float, u);
}

// ---------------------------------------------------------------------------
// 1) Gene-group block-diagonal projection + ReLU. Streams 200MB of x.
//    One thread per (batch, group); a wave covers 32 consecutive groups, so
//    the wave collectively touches one contiguous 3200B span of x per row.
// ---------------------------------------------------------------------------
__global__ void __launch_bounds__(256)
gene_kernel(const float* __restrict__ x, const float* __restrict__ gene_w,
            const float* __restrict__ gene_b,
            uint16_t* __restrict__ g_bf, float* __restrict__ g_f32) {
  int tid = blockIdx.x * blockDim.x + threadIdx.x;          // over kB * kGP
  if (tid >= kB * kGP) return;
  int b  = tid / kGP;
  int gr = tid % kGP;
  float acc = 0.0f;
  if (gr < kG) {
    const float* xr = x + (size_t)b * (kG * kGS) + (size_t)gr * kGS;
    const float* wr = gene_w + (size_t)gr * kGS;
#pragma unroll
    for (int k = 0; k < kGS; ++k) acc = fmaf(xr[k], wr[k], acc);
    acc += gene_b[gr];
    acc = fmaxf(acc, 0.0f);
    g_f32[(size_t)b * kG + gr] = acc;                       // fp32 for residual head
  }
  g_bf[(size_t)b * kGP + gr] = f2bf(acc);                   // bf16 (zero-padded) for WMMA
}

// ---------------------------------------------------------------------------
// 2) Weight transpose + bf16 convert: W [K,N] row-major -> Wt [N,Kp] row-major
//    (zero-pad K to Kp). Gives contiguous per-lane B-fragment loads.
// ---------------------------------------------------------------------------
__global__ void __launch_bounds__(256)
conv_transpose_bf16(const float* __restrict__ W, uint16_t* __restrict__ Wt,
                    int K, int N, int Kp) {
  int tid = blockIdx.x * blockDim.x + threadIdx.x;          // over N * Kp
  if (tid >= N * Kp) return;
  int n = tid / Kp;
  int k = tid % Kp;
  float v = (k < K) ? W[(size_t)k * N + n] : 0.0f;
  Wt[tid] = f2bf(v);
}

// ---------------------------------------------------------------------------
// 3) bf16 WMMA GEMM:  C[M,N] = A[M,Kp] * Bt[N,Kp]^T  (fp32 accumulate)
//    One wave owns a 32x32 output tile (2x2 v_wmma_f32_16x16x32_bf16).
//    A-fragment (16x32 bf16 ISA layout): lane l<16 holds row m=l,
//      K = kb+8*half .. +7 (VGPR0-3) and kb+16+8*half .. (VGPR4-7)
//      -> two contiguous 16B loads per lane.
//    B-fragment (32x16 bf16): lane l holds col n=l, same K order as A:
//      kb+8*half .. +7 and kb+16+8*half .. -> two contiguous 16B loads per lane.
// ---------------------------------------------------------------------------
__global__ void __launch_bounds__(256)
wmma_gemm_bf16(const uint16_t* __restrict__ A, const uint16_t* __restrict__ Bt,
               float* __restrict__ C, int M, int N, int Kp) {
  const int lane = threadIdx.x & 31;
  const int wave = threadIdx.x >> 5;
  const int tilesN = N >> 5;
  const int tile = blockIdx.x * 8 + wave;
  if (tile >= (M >> 5) * tilesN) return;
  const int tm = (tile / tilesN) << 5;
  const int tn = (tile % tilesN) << 5;
  const int half = lane >> 4;
  const int l    = lane & 15;

  v8f acc00 = {}, acc01 = {}, acc10 = {}, acc11 = {};

  const uint16_t* a0p = A  + (size_t)(tm + l)      * Kp + 8  * half;
  const uint16_t* a1p = A  + (size_t)(tm + 16 + l) * Kp + 8  * half;
  const uint16_t* b0p = Bt + (size_t)(tn + l)      * Kp + 8  * half;
  const uint16_t* b1p = Bt + (size_t)(tn + 16 + l) * Kp + 8  * half;

  for (int k = 0; k < Kp; k += 32) {
    FragAB a0, a1, b0, b1;
    a0.f[0] = *(const f32x4*)(a0p + k);
    a0.f[1] = *(const f32x4*)(a0p + k + 16);
    a1.f[0] = *(const f32x4*)(a1p + k);
    a1.f[1] = *(const f32x4*)(a1p + k + 16);
    b0.f[0] = *(const f32x4*)(b0p + k);
    b0.f[1] = *(const f32x4*)(b0p + k + 16);
    b1.f[0] = *(const f32x4*)(b1p + k);
    b1.f[1] = *(const f32x4*)(b1p + k + 16);
    acc00 = __builtin_amdgcn_wmma_f32_16x16x32_bf16(false, a0.v, false, b0.v,
                                                    (short)0, acc00, false, false);
    acc01 = __builtin_amdgcn_wmma_f32_16x16x32_bf16(false, a0.v, false, b1.v,
                                                    (short)0, acc01, false, false);
    acc10 = __builtin_amdgcn_wmma_f32_16x16x32_bf16(false, a1.v, false, b0.v,
                                                    (short)0, acc10, false, false);
    acc11 = __builtin_amdgcn_wmma_f32_16x16x32_bf16(false, a1.v, false, b1.v,
                                                    (short)0, acc11, false, false);
  }

  // C/D layout: lanes 0-15 -> M = j, lanes 16-31 -> M = 8+j; N = lane&15
#pragma unroll
  for (int j = 0; j < 8; ++j) {
    int r0 = tm + 8 * half + j;
    int r1 = r0 + 16;
    C[(size_t)r0 * N + tn + l]      = acc00[j];
    C[(size_t)r0 * N + tn + 16 + l] = acc01[j];
    C[(size_t)r1 * N + tn + l]      = acc10[j];
    C[(size_t)r1 * N + tn + 16 + l] = acc11[j];
  }
}

// ---------------------------------------------------------------------------
// 4) Training-mode BN folded to per-column affine: a=gamma*rsqrt(var+eps),
//    s=beta-mu*a. One thread per column; column-loop reads are coalesced.
//    (Pre-BN bias cancels in h-mu, so b1/b2 are never needed.)
// ---------------------------------------------------------------------------
__global__ void __launch_bounds__(256)
bn_stats(const float* __restrict__ H, int Bn, int N,
         const float* __restrict__ gamma, const float* __restrict__ beta,
         float* __restrict__ a, float* __restrict__ s) {
  int c = blockIdx.x * blockDim.x + threadIdx.x;
  if (c >= N) return;
  float sum = 0.0f, sq = 0.0f;
  for (int r = 0; r < Bn; ++r) {
    float v = H[(size_t)r * N + c];
    sum += v;
    sq = fmaf(v, v, sq);
  }
  float mu  = sum / (float)Bn;
  float var = sq / (float)Bn - mu * mu;
  float ai  = gamma[c] * rsqrtf(var + kEps);
  a[c] = ai;
  s[c] = beta[c] - mu * ai;
}

__global__ void __launch_bounds__(256)
bn_relu_bf16(const float* __restrict__ H, const float* __restrict__ a,
             const float* __restrict__ s, uint16_t* __restrict__ outbf,
             int total, int N) {
  int tid = blockIdx.x * blockDim.x + threadIdx.x;
  if (tid >= total) return;
  int c = tid % N;
  float v = fmaxf(fmaf(a[c], H[tid], s[c]), 0.0f);
  outbf[tid] = f2bf(v);
}

// ---------------------------------------------------------------------------
// 5) Head: logits = h2@Wout + bout + g@Wres + bres; softmax over 10 classes.
//    One block per batch row; block-wide LDS reduction per class.
// ---------------------------------------------------------------------------
__global__ void __launch_bounds__(256)
head_softmax(const uint16_t* __restrict__ h2bf, const float* __restrict__ gf,
             const float* __restrict__ Wout, const float* __restrict__ bout,
             const float* __restrict__ Wres, const float* __restrict__ bres,
             float* __restrict__ out) {
  __shared__ float red[256];
  __shared__ float logits[kC];
  const int b = blockIdx.x;
  const int t = threadIdx.x;

  float acc[kC];
#pragma unroll
  for (int c = 0; c < kC; ++c) acc[c] = 0.0f;

  const uint16_t* hrow = h2bf + (size_t)b * kH2;
  for (int k = t; k < kH2; k += 256) {
    float h = bf2f(hrow[k]);
    const float* wr = Wout + (size_t)k * kC;
#pragma unroll
    for (int c = 0; c < kC; ++c) acc[c] = fmaf(h, wr[c], acc[c]);
  }
  const float* grow = gf + (size_t)b * kG;
  for (int k = t; k < kG; k += 256) {
    float gv = grow[k];
    const float* wr = Wres + (size_t)k * kC;
#pragma unroll
    for (int c = 0; c < kC; ++c) acc[c] = fmaf(gv, wr[c], acc[c]);
  }

  for (int c = 0; c < kC; ++c) {
    red[t] = acc[c];
    __syncthreads();
    for (int off = 128; off > 0; off >>= 1) {
      if (t < off) red[t] += red[t + off];
      __syncthreads();
    }
    if (t == 0) logits[c] = red[0] + bout[c] + bres[c];
    __syncthreads();
  }

  if (t == 0) {
    float m = -1e30f;
#pragma unroll
    for (int c = 0; c < kC; ++c) m = fmaxf(m, logits[c]);
    float e[kC];
    float sum = 0.0f;
#pragma unroll
    for (int c = 0; c < kC; ++c) { e[c] = __expf(logits[c] - m); sum += e[c]; }
    float inv = 1.0f / sum;
#pragma unroll
    for (int c = 0; c < kC; ++c) out[(size_t)b * kC + c] = e[c] * inv;
  }
}

// ---------------------------------------------------------------------------
extern "C" void kernel_launch(void* const* d_in, const int* in_sizes, int n_in,
                              void* d_out, int out_size, void* d_ws, size_t ws_size,
                              hipStream_t stream) {
  (void)in_sizes; (void)n_in; (void)out_size; (void)ws_size;
  const float* x      = (const float*)d_in[0];
  const float* gene_w = (const float*)d_in[1];
  const float* gene_b = (const float*)d_in[2];
  const float* W1     = (const float*)d_in[3];
  /* b1 = d_in[4]: cancels inside training-mode BN */
  const float* gamma1 = (const float*)d_in[5];
  const float* beta1  = (const float*)d_in[6];
  const float* W2     = (const float*)d_in[7];
  /* b2 = d_in[8]: cancels inside training-mode BN */
  const float* gamma2 = (const float*)d_in[9];
  const float* beta2  = (const float*)d_in[10];
  const float* Wout   = (const float*)d_in[11];
  const float* bout   = (const float*)d_in[12];
  const float* Wres   = (const float*)d_in[13];
  const float* bres   = (const float*)d_in[14];
  float* out = (float*)d_out;

  // ---- workspace carve-up (all offsets 256B-aligned) ----
  char* p = (char*)d_ws;
  size_t off = 0;
  auto take = [&](size_t bytes) {
    void* r = p + off;
    off += (bytes + 255) & ~(size_t)255;
    return r;
  };
  uint16_t* g_bf  = (uint16_t*)take((size_t)kB * kGP * 2);   // 4 MB
  float*    g_f32 = (float*)   take((size_t)kB * kG  * 4);   // 8 MB
  uint16_t* W1t   = (uint16_t*)take((size_t)kH1 * kGP * 2);  // 4 MB
  float*    h1pre = (float*)   take((size_t)kB * kH1 * 4);   // 4 MB
  float*    a1    = (float*)   take((size_t)kH1 * 4);
  float*    s1    = (float*)   take((size_t)kH1 * 4);
  uint16_t* h1bf  = (uint16_t*)take((size_t)kB * kH1 * 2);   // 2 MB
  uint16_t* W2t   = (uint16_t*)take((size_t)kH2 * kH1 * 2);  // 1 MB
  float*    h2pre = (float*)   take((size_t)kB * kH2 * 4);   // 2 MB
  float*    a2    = (float*)   take((size_t)kH2 * 4);
  float*    s2    = (float*)   take((size_t)kH2 * 4);
  uint16_t* h2bf  = (uint16_t*)take((size_t)kB * kH2 * 2);   // 1 MB

  // 1) gene projection (streams 200MB of x; the roofline floor)
  gene_kernel<<<(kB * kGP) / 256, 256, 0, stream>>>(x, gene_w, gene_b, g_bf, g_f32);

  // 2) weight transpose+convert (L2-resident, tiny)
  conv_transpose_bf16<<<(kH1 * kGP) / 256, 256, 0, stream>>>(W1, W1t, kG, kH1, kGP);
  conv_transpose_bf16<<<(kH2 * kH1) / 256, 256, 0, stream>>>(W2, W2t, kH1, kH2, kH1);

  // 3) GEMM1: [1024,2048]x[2048,1024] -> 1024 wave-tiles -> 128 blocks of 8 waves
  wmma_gemm_bf16<<<128, 256, 0, stream>>>(g_bf, W1t, h1pre, kB, kH1, kGP);
  bn_stats<<<kH1 / 256, 256, 0, stream>>>(h1pre, kB, kH1, gamma1, beta1, a1, s1);
  bn_relu_bf16<<<(kB * kH1) / 256, 256, 0, stream>>>(h1pre, a1, s1, h1bf, kB * kH1, kH1);

  // 4) GEMM2: [1024,1024]x[1024,512] -> 512 wave-tiles -> 64 blocks
  wmma_gemm_bf16<<<64, 256, 0, stream>>>(h1bf, W2t, h2pre, kB, kH2, kH1);
  bn_stats<<<kH2 / 256, 256, 0, stream>>>(h2pre, kB, kH2, gamma2, beta2, a2, s2);
  bn_relu_bf16<<<(kB * kH2) / 256, 256, 0, stream>>>(h2pre, a2, s2, h2bf, kB * kH2, kH2);

  // 5) heads + softmax
  head_softmax<<<kB, 256, 0, stream>>>(h2bf, g_f32, Wout, bout, Wres, bres, out);
}
